// GatedDeltaLayer_67989332296149
// MI455X (gfx1250) — matrix operand order repaired and checked by reading, not compiled, not run on hardware
//
#include <hip/hip_runtime.h>
#include <hip/hip_bf16.h>
#include <math.h>

#define Bn   2
#define Ln   2048
#define DM   1024
#define Hh   4
#define DH   128
#define TOT  512
#define Mrows (Bn*Ln)   // 4096

typedef __bf16 bf16_t;
typedef bf16_t v16bf __attribute__((ext_vector_type(16)));
typedef float  v8f   __attribute__((ext_vector_type(8)));

struct alignas(16) u128 { unsigned int x[4]; };
union Frag { u128 u[2]; v16bf v; };

__device__ __forceinline__ unsigned short f2bf(float f) {
    unsigned int u = __float_as_uint(f);
    unsigned int r = u + 0x7FFFu + ((u >> 16) & 1u);
    return (unsigned short)(r >> 16);
}
__device__ __forceinline__ float siluf(float x) {
    return x * (1.0f / (1.0f + __expf(-x)));
}

// ---------------------------------------------------------------- f32 -> bf16
__global__ __launch_bounds__(256) void cvt_bf16_kernel(const float* __restrict__ src,
                                                       unsigned short* __restrict__ dst,
                                                       size_t n) {
    size_t i = (size_t)blockIdx.x * 256 + threadIdx.x;
    if (i < n) dst[i] = f2bf(src[i]);
}

// ------------------------------------------------- WMMA GEMM: C = A(MxK) * Wt(NxK)^T
// bf16 inputs (ushort-packed), fp32 out. M%64==0, N%64==0, K%32==0.
// block = 128 threads (4 waves); wave -> 16(M) x 64(N); block -> 64 x 64.
__global__ __launch_bounds__(128) void wmma_gemm_bf16_kernel(const unsigned short* __restrict__ A,
                                                             const unsigned short* __restrict__ Wt,
                                                             float* __restrict__ C,
                                                             int M, int N, int K) {
    const int lane = threadIdx.x & 31;
    const int wave = threadIdx.x >> 5;
    const int l15  = lane & 15;
    const int hi   = lane >> 4;             // 0 for lanes 0-15, 1 for 16-31
    const int m0   = blockIdx.y * 64 + wave * 16;
    const int n0   = blockIdx.x * 64;

    const unsigned short* arow = A + (size_t)(m0 + l15) * K;
    const unsigned short* br0  = Wt + (size_t)(n0 +  0 + l15) * K;
    const unsigned short* br1  = Wt + (size_t)(n0 + 16 + l15) * K;
    const unsigned short* br2  = Wt + (size_t)(n0 + 32 + l15) * K;
    const unsigned short* br3  = Wt + (size_t)(n0 + 48 + l15) * K;

    const int aOff = hi ? 16 : 0;   // A: lanes 16-31 hold K 16..31
    const int bOff = hi ? 16 : 0;   // B: lanes 16-31 hold K 16..31

    v8f acc0 = {}, acc1 = {}, acc2 = {}, acc3 = {};

    for (int k0 = 0; k0 < K; k0 += 32) {
        Frag af;
        af.u[0] = *reinterpret_cast<const u128*>(arow + k0 + aOff);       // K off..off+7
        af.u[1] = *reinterpret_cast<const u128*>(arow + k0 + aOff + 8);   // K off+8..off+15
        Frag bf0, bf1, bf2, bf3;
        bf0.u[0] = *reinterpret_cast<const u128*>(br0 + k0 + bOff);
        bf0.u[1] = *reinterpret_cast<const u128*>(br0 + k0 + bOff + 8);
        bf1.u[0] = *reinterpret_cast<const u128*>(br1 + k0 + bOff);
        bf1.u[1] = *reinterpret_cast<const u128*>(br1 + k0 + bOff + 8);
        bf2.u[0] = *reinterpret_cast<const u128*>(br2 + k0 + bOff);
        bf2.u[1] = *reinterpret_cast<const u128*>(br2 + k0 + bOff + 8);
        bf3.u[0] = *reinterpret_cast<const u128*>(br3 + k0 + bOff);
        bf3.u[1] = *reinterpret_cast<const u128*>(br3 + k0 + bOff + 8);

        acc0 = __builtin_amdgcn_wmma_f32_16x16x32_bf16(false, af.v, false, bf0.v, (short)0, acc0, false, false);
        acc1 = __builtin_amdgcn_wmma_f32_16x16x32_bf16(false, af.v, false, bf1.v, (short)0, acc1, false, false);
        acc2 = __builtin_amdgcn_wmma_f32_16x16x32_bf16(false, af.v, false, bf2.v, (short)0, acc2, false, false);
        acc3 = __builtin_amdgcn_wmma_f32_16x16x32_bf16(false, af.v, false, bf3.v, (short)0, acc3, false, false);
    }

    const int rowb = m0 + (hi ? 8 : 0);
#pragma unroll
    for (int r = 0; r < 8; ++r) {
        size_t rbase = (size_t)(rowb + r) * N;
        C[rbase + n0 +  0 + l15] = acc0[r];
        C[rbase + n0 + 16 + l15] = acc1[r];
        C[rbase + n0 + 32 + l15] = acc2[r];
        C[rbase + n0 + 48 + l15] = acc3[r];
    }
}

// ----------------------------------------------- alpha/beta: sigmoid(x @ W^T), N=4 each
__global__ __launch_bounds__(256) void ab_kernel(const float* __restrict__ x,
                                                 const float* __restrict__ Wa,
                                                 const float* __restrict__ Wb,
                                                 float* __restrict__ alpha,
                                                 float* __restrict__ beta) {
    const int m    = blockIdx.x;
    const int grp  = threadIdx.x >> 5;  // 0..7 (4 alpha heads, 4 beta heads)
    const int lane = threadIdx.x & 31;
    const float* w = (grp < Hh) ? (Wa + (size_t)grp * DM) : (Wb + (size_t)(grp - Hh) * DM);
    const float* xr = x + (size_t)m * DM;
    float s = 0.f;
    for (int k = lane; k < DM; k += 32) s += xr[k] * w[k];
#pragma unroll
    for (int off = 16; off > 0; off >>= 1) s += __shfl_down(s, off, 32);
    if (lane == 0) {
        float sg = 1.0f / (1.0f + __expf(-s));
        if (grp < Hh) alpha[(size_t)m * Hh + grp] = sg;
        else          beta [(size_t)m * Hh + grp - Hh] = sg;
    }
}

// ------------------------------------- causal depthwise conv (width 4, pad 3 trimmed) + SiLU
__global__ __launch_bounds__(256) void dwconv_silu_kernel(const float* __restrict__ xin,
                                                          const float* __restrict__ w,
                                                          const float* __restrict__ bias,
                                                          float* __restrict__ out,
                                                          int C) {
    size_t idx = (size_t)blockIdx.x * 256 + threadIdx.x;
    size_t n = (size_t)Bn * Ln * C;
    if (idx >= n) return;
    int c = (int)(idx % C);
    size_t bl = idx / C;
    int l = (int)(bl % Ln);
    int b = (int)(bl / Ln);
    const float* xp = xin + ((size_t)b * Ln) * C + c;
    const float* wt = w + (size_t)c * 4;
    float acc = bias[c];
#pragma unroll
    for (int tau = 0; tau < 4; ++tau) {
        int ll = l - 3 + tau;
        if (ll >= 0) acc += wt[tau] * xp[(size_t)ll * C];
    }
    out[idx] = siluf(acc);
}

// ----------------------------------------------- L2 normalize over 512 dims, in place
__global__ __launch_bounds__(256) void l2norm512_kernel(float* __restrict__ buf) {
    const int m = blockIdx.x;
    const int t = threadIdx.x;
    float* row = buf + (size_t)m * TOT;
    float a0 = row[t], a1 = row[t + 256];
    __shared__ float red[256];
    red[t] = a0 * a0 + a1 * a1;
    __syncthreads();
#pragma unroll
    for (int s = 128; s > 0; s >>= 1) {
        if (t < s) red[t] += red[t + s];
        __syncthreads();
    }
    float sc = 1.0f / fmaxf(sqrtf(red[0]), 1e-12f);
    row[t] = a0 * sc;
    row[t + 256] = a1 * sc;
}

// ----------------------------------------------- sequential gated-delta scan
// grid = B*H blocks, 128 threads; thread i owns state row S[i, 0..127] in VGPRs.
// Double-buffered: global loads for step t+1 are issued before the compute of step t,
// so their latency hides behind ~512 VALU ops. k/q broadcast via LDS read as float4.
__global__ __launch_bounds__(128) void scan_kernel(const float* __restrict__ qc,
                                                   const float* __restrict__ kc,
                                                   const float* __restrict__ vc,
                                                   const float* __restrict__ alpha,
                                                   const float* __restrict__ beta,
                                                   const float* __restrict__ state0,
                                                   float* __restrict__ obuf,
                                                   float* __restrict__ state_out) {
    const int bh = blockIdx.x;
    const int b = bh / Hh, h = bh % Hh;
    const int i = threadIdx.x;

    float S[DH];
    const float* s0 = state0 + ((size_t)(b * Hh + h) * DH + i) * DH;
#pragma unroll
    for (int j = 0; j < DH; ++j) S[j] = s0[j];

    __shared__ alignas(16) float kk[DH];
    __shared__ alignas(16) float qq[DH];

    // prefetch step 0
    size_t base  = ((size_t)b * Ln) * TOT + (size_t)h * DH + i;
    size_t vbase = ((size_t)b * Ln) * (2 * TOT) + (size_t)h * DH + i;
    size_t abase = ((size_t)b * Ln) * Hh + h;
    float nk  = kc[base];
    float nq  = qc[base];
    float nv  = vc[vbase];
    float nvg = vc[vbase + TOT];
    float na  = alpha[abase];
    float nb  = beta[abase];

    for (int t = 0; t < Ln; ++t) {
        kk[i] = nk;
        qq[i] = nq;
        const float vi = nv, vg = nvg, a = na, bb = nb;
        __syncthreads();

        // issue next step's loads so they overlap the compute below
        if (t + 1 < Ln) {
            size_t nbase  = base  + TOT;
            size_t nvb    = vbase + 2 * TOT;
            size_t nab    = abase + Hh;
            nk  = kc[nbase];
            nq  = qc[nbase];
            nv  = vc[nvb];
            nvg = vc[nvb + TOT];
            na  = alpha[nab];
            nb  = beta[nab];
        }

        const float4* k4 = reinterpret_cast<const float4*>(kk);
        const float4* q4 = reinterpret_cast<const float4*>(qq);

        float Sk = 0.f;
#pragma unroll
        for (int j4 = 0; j4 < DH / 4; ++j4) {
            float4 kv = k4[j4];
            Sk = fmaf(S[4 * j4 + 0], kv.x, Sk);
            Sk = fmaf(S[4 * j4 + 1], kv.y, Sk);
            Sk = fmaf(S[4 * j4 + 2], kv.z, Sk);
            Sk = fmaf(S[4 * j4 + 3], kv.w, Sk);
        }
        const float wc = bb * vi - a * bb * Sk;   // S = a*S + (b*v_i - a*b*Sk_i)*k_j
        float o = 0.f;
#pragma unroll
        for (int j4 = 0; j4 < DH / 4; ++j4) {
            float4 kv = k4[j4];
            float4 qv = q4[j4];
            S[4 * j4 + 0] = fmaf(a, S[4 * j4 + 0], wc * kv.x);
            S[4 * j4 + 1] = fmaf(a, S[4 * j4 + 1], wc * kv.y);
            S[4 * j4 + 2] = fmaf(a, S[4 * j4 + 2], wc * kv.z);
            S[4 * j4 + 3] = fmaf(a, S[4 * j4 + 3], wc * kv.w);
            o = fmaf(S[4 * j4 + 0], qv.x, o);
            o = fmaf(S[4 * j4 + 1], qv.y, o);
            o = fmaf(S[4 * j4 + 2], qv.z, o);
            o = fmaf(S[4 * j4 + 3], qv.w, o);
        }
        obuf[base] = o * vg;

        base  += TOT;
        vbase += 2 * TOT;
        abase += Hh;
        __syncthreads();
    }

    float* so = state_out + ((size_t)(b * Hh + h) * DH + i) * DH;
#pragma unroll
    for (int j = 0; j < DH; ++j) so[j] = S[j];
}

// --------------------------------- out = LN(o * silu(g)) * ln_w + ln_b, emitted as bf16
__global__ __launch_bounds__(256) void gate_ln_kernel(const float* __restrict__ obuf,
                                                      const float* __restrict__ gpre,
                                                      const float* __restrict__ lnw,
                                                      const float* __restrict__ lnb,
                                                      unsigned short* __restrict__ ybf) {
    const int m = blockIdx.x;
    const int t = threadIdx.x;
    size_t base = (size_t)m * TOT;
    float x0 = obuf[base + t]       * siluf(gpre[base + t]);
    float x1 = obuf[base + t + 256] * siluf(gpre[base + t + 256]);
    __shared__ float red[256];
    red[t] = x0 + x1;
    __syncthreads();
#pragma unroll
    for (int s = 128; s > 0; s >>= 1) { if (t < s) red[t] += red[t + s]; __syncthreads(); }
    float mu = red[0] * (1.0f / TOT);
    __syncthreads();
    float d0 = x0 - mu, d1 = x1 - mu;
    red[t] = d0 * d0 + d1 * d1;
    __syncthreads();
#pragma unroll
    for (int s = 128; s > 0; s >>= 1) { if (t < s) red[t] += red[t + s]; __syncthreads(); }
    float rs = rsqrtf(red[0] * (1.0f / TOT) + 1e-5f);
    ybf[base + t]       = f2bf(d0 * rs * lnw[t]       + lnb[t]);
    ybf[base + t + 256] = f2bf(d1 * rs * lnw[t + 256] + lnb[t + 256]);
}

// ======================================================================= host
extern "C" void kernel_launch(void* const* d_in, const int* in_sizes, int n_in,
                              void* d_out, int out_size, void* d_ws, size_t ws_size,
                              hipStream_t stream) {
    (void)in_sizes; (void)n_in; (void)out_size; (void)ws_size;
    const float* x     = (const float*)d_in[0];
    const float* state = (const float*)d_in[1];
    const float* Wq    = (const float*)d_in[2];
    const float* Wk    = (const float*)d_in[3];
    const float* Wv    = (const float*)d_in[4];
    const float* Wa    = (const float*)d_in[5];
    const float* Wb    = (const float*)d_in[6];
    const float* Wg    = (const float*)d_in[7];
    const float* Wo    = (const float*)d_in[8];
    const float* qcw   = (const float*)d_in[9];
    const float* qcb   = (const float*)d_in[10];
    const float* kcw   = (const float*)d_in[11];
    const float* kcb   = (const float*)d_in[12];
    const float* vcw   = (const float*)d_in[13];
    const float* vcb   = (const float*)d_in[14];
    const float* lnw   = (const float*)d_in[15];
    const float* lnb   = (const float*)d_in[16];

    float* out_main  = (float*)d_out;                              // (B,L,DM)
    float* out_state = (float*)d_out + (size_t)Bn * Ln * DM;       // (B,H,DH,DH)

    // workspace carve-out
    char* w = (char*)d_ws;
    size_t off = 0;
    auto alloc = [&](size_t bytes) -> void* {
        void* p = w + off;
        off = (off + bytes + 255) & ~(size_t)255;
        return p;
    };
    unsigned short* xbf  = (unsigned short*)alloc((size_t)Mrows * DM * 2);
    unsigned short* wqb  = (unsigned short*)alloc((size_t)TOT * DM * 2);
    unsigned short* wkb  = (unsigned short*)alloc((size_t)TOT * DM * 2);
    unsigned short* wvb  = (unsigned short*)alloc((size_t)2 * TOT * DM * 2);
    unsigned short* wgb  = (unsigned short*)alloc((size_t)TOT * DM * 2);
    unsigned short* wob  = (unsigned short*)alloc((size_t)DM * TOT * 2);
    float* qpre  = (float*)alloc((size_t)Mrows * TOT * 4);
    float* kpre  = (float*)alloc((size_t)Mrows * TOT * 4);
    float* vpre  = (float*)alloc((size_t)Mrows * 2 * TOT * 4);
    float* gpre  = (float*)alloc((size_t)Mrows * TOT * 4);
    float* qc    = (float*)alloc((size_t)Mrows * TOT * 4);
    float* kc    = (float*)alloc((size_t)Mrows * TOT * 4);
    float* vc    = (float*)alloc((size_t)Mrows * 2 * TOT * 4);
    float* alpha = (float*)alloc((size_t)Mrows * Hh * 4);
    float* beta  = (float*)alloc((size_t)Mrows * Hh * 4);
    float* obuf  = (float*)alloc((size_t)Mrows * TOT * 4);
    unsigned short* ybf = (unsigned short*)alloc((size_t)Mrows * TOT * 2);

    auto cvt = [&](const float* s, unsigned short* d, size_t n) {
        cvt_bf16_kernel<<<dim3((unsigned)((n + 255) / 256)), dim3(256), 0, stream>>>(s, d, n);
    };
    cvt(x,  xbf, (size_t)Mrows * DM);
    cvt(Wq, wqb, (size_t)TOT * DM);
    cvt(Wk, wkb, (size_t)TOT * DM);
    cvt(Wv, wvb, (size_t)2 * TOT * DM);
    cvt(Wg, wgb, (size_t)TOT * DM);
    cvt(Wo, wob, (size_t)DM * TOT);

    // projections: C(M x N) = Xbf(M x K) * Wt(N x K)^T
    auto gemm = [&](const unsigned short* A, const unsigned short* Wt, float* C,
                    int M, int N, int K) {
        wmma_gemm_bf16_kernel<<<dim3(N / 64, M / 64), dim3(128), 0, stream>>>(A, Wt, C, M, N, K);
    };
    gemm(xbf, wqb, qpre, Mrows, TOT, DM);
    gemm(xbf, wkb, kpre, Mrows, TOT, DM);
    gemm(xbf, wvb, vpre, Mrows, 2 * TOT, DM);
    gemm(xbf, wgb, gpre, Mrows, TOT, DM);

    ab_kernel<<<dim3(Mrows), dim3(256), 0, stream>>>(x, Wa, Wb, alpha, beta);

    {   // depthwise conv + SiLU
        size_t nq = (size_t)Mrows * TOT;
        size_t nv = (size_t)Mrows * 2 * TOT;
        dwconv_silu_kernel<<<dim3((unsigned)((nq + 255) / 256)), dim3(256), 0, stream>>>(qpre, qcw, qcb, qc, TOT);
        dwconv_silu_kernel<<<dim3((unsigned)((nq + 255) / 256)), dim3(256), 0, stream>>>(kpre, kcw, kcb, kc, TOT);
        dwconv_silu_kernel<<<dim3((unsigned)((nv + 255) / 256)), dim3(256), 0, stream>>>(vpre, vcw, vcb, vc, 2 * TOT);
    }

    l2norm512_kernel<<<dim3(Mrows), dim3(256), 0, stream>>>(qc);
    l2norm512_kernel<<<dim3(Mrows), dim3(256), 0, stream>>>(kc);

    scan_kernel<<<dim3(Bn * Hh), dim3(128), 0, stream>>>(qc, kc, vc, alpha, beta, state,
                                                         obuf, out_state);

    gate_ln_kernel<<<dim3(Mrows), dim3(256), 0, stream>>>(obuf, gpre, lnw, lnb, ybf);

    // final projection: out = y(4096x512) @ Wo^T(512x1024) -> (4096x1024)
    gemm(ybf, wob, out_main, Mrows, DM, TOT);
}
